// NonLocalKmeansAttention_6133213299489
// MI455X (gfx1250) — matrix operand order repaired and checked by reading, not compiled, not run on hardware
//
#include <hip/hip_runtime.h>
#include <hip/hip_bf16.h>

typedef _Float16 half_t;
typedef __attribute__((ext_vector_type(16))) _Float16 v16h;
typedef __attribute__((ext_vector_type(8)))  _Float16 v8h;
typedef __attribute__((ext_vector_type(8)))  float    v8f;
typedef int vs4i __attribute__((__vector_size__(16)));   // matches builtin proto

#define N_B     8
#define C_IN    256
#define C_R     64
#define H_IMG   96
#define W_IMG   96
#define L_PIX   (H_IMG * W_IMG)     // 9216
#define WIN     144
#define KWIN    (L_PIX / WIN)       // 64 windows, padding == 0
#define NCLUST  128
#define K3      (C_IN * 9)          // 2304
#define NCHUNK  (L_PIX / 256)       // 36
#define ATTN_THREADS 288            // 9 waves (wave32): 144 rows / 16 per wave

// ---------------- CDNA5 async global->LDS staging (ASYNCcnt) ------------------
// GLOBAL_LOAD_ASYNC_TO_LDS_B128: (v4i addrspace(1)*, v4i addrspace(3)*, imm, imm)

#define AS1 __attribute__((address_space(1)))
#define AS3 __attribute__((address_space(3)))

#if defined(__HIP_DEVICE_COMPILE__) &&                                         \
    __has_builtin(__builtin_amdgcn_global_load_async_to_lds_b128) &&           \
    __has_builtin(__builtin_amdgcn_s_wait_asynccnt)
#define HAVE_ASYNC 1
#else
#define HAVE_ASYNC 0
#endif

__device__ inline void async_cp16(void* lds_dst, const void* gsrc) {
#if HAVE_ASYNC
  __builtin_amdgcn_global_load_async_to_lds_b128((AS1 vs4i*)gsrc,
                                                 (AS3 vs4i*)lds_dst, 0, 0);
#else
  *(uint4*)lds_dst = *(const uint4*)gsrc;
#endif
}

__device__ inline void async_wait_all() {
#if HAVE_ASYNC
  __builtin_amdgcn_s_wait_asynccnt(0);
#endif
}

// ---------------- WMMA fragment helpers (contiguous K order, same for A and B) ----

// A fragment: 16x32 f16, row-major source, row = lane&15,
// K slots: idx -> kh*16+idx, kh = lane>>4
__device__ inline v16h ld_frag_a(const half_t* base, int ld, int lane) {
  const half_t* p = base + (lane & 15) * ld + ((lane >> 4) << 4);
  v16h f;
#pragma unroll
  for (int i = 0; i < 16; ++i) f[i] = p[i];
  return f;
}

// 16 contiguous halves per lane (fragment-ready swizzled storage -> 2x b128)
__device__ inline v16h ld_frag_ctg(const half_t* p) {
  v16h f;
#pragma unroll
  for (int i = 0; i < 16; ++i) f[i] = p[i];
  return f;
}

// B fragment when B is stored transposed (n-major, i.e. keys[j][d]): contiguous K
__device__ inline v16h ld_frag_bT(const half_t* base, int ldn, int lane) {
  return ld_frag_ctg(base + (lane & 15) * ldn + ((lane >> 4) << 4));
}

__device__ inline v8f wmma32(v16h a, v16h b, v8f c) {
  return __builtin_amdgcn_wmma_f32_16x16x32_f16(false, a, false, b, (short)0, c,
                                                false, false);
}

// ---------------- kernel 0: weights -> f16 fragment-ready swizzled blocks -----
// per (ktile32, ntile16) block of 512 halves: [g = k>>4 & 1][n-lane][16 k-slots]

__global__ __launch_bounds__(256) void prep_weights(const float* __restrict__ wm,
                                                    const float* __restrict__ wa,
                                                    half_t* __restrict__ B3,
                                                    half_t* __restrict__ B1) {
  int idx = blockIdx.x * 256 + threadIdx.x;
  if (idx < K3 * C_R) {
    int k = idx / C_R, o = idx - k * C_R;
    int kt = k >> 5, g = (k >> 4) & 1, i = k & 15;
    int nt = o >> 4, nl = o & 15;
    size_t dst = ((size_t)(kt * 4 + nt) * 2 + g) * 256 + nl * 16 + i;
    B3[dst] = (half_t)wm[o * K3 + k];
  }
  int idx2 = idx - K3 * C_R;
  if (idx2 >= 0 && idx2 < C_IN * C_IN) {
    int k = idx2 / C_IN, o = idx2 - k * C_IN;
    int kt = k >> 5, g = (k >> 4) & 1, i = k & 15;
    int nt = o >> 4, nl = o & 15;
    size_t dst = ((size_t)(kt * 16 + nt) * 2 + g) * 256 + nl * 16 + i;
    B1[dst] = (half_t)wa[o * C_IN + k];
  }
}

// ---------------- kernel 1: 3x3 conv as implicit GEMM (WMMA) ------------------
// per WG: 128 pixels x 64 out-channels, K = 2304 in 32-steps. 8 waves, 16 rows each.

__global__ __launch_bounds__(256) void conv3x3_wmma(const float* __restrict__ x,
                                                    const half_t* __restrict__ B3w,
                                                    const float* __restrict__ bias,
                                                    float* __restrict__ xemb) {
  __shared__ half_t As[128 * 32];   // [m][kk]
  __shared__ half_t Bs[4 * 512];    // 4 swizzled (32k x 16n) blocks
  __shared__ int kdec[K3];          // k -> (c<<4)|(ky<<2)|kx, built once
  int n  = blockIdx.x / (L_PIX / 128);
  int l0 = (blockIdx.x % (L_PIX / 128)) * 128;
  int t = threadIdx.x, wave = t >> 5, lane = t & 31;
  const float* xin = x + (size_t)n * C_IN * L_PIX;

  for (int e = t; e < K3; e += 256) {
    int c = e / 9, r = e - c * 9;
    kdec[e] = (c << 4) | ((r / 3) << 2) | (r - (r / 3) * 3);
  }

  // per-thread staging coords are iteration-invariant: m = t&127
  int m = t & 127, kkb = t >> 7;
  int lq = l0 + m, hq = lq / W_IMG, wq = lq - hq * W_IMG;
  int fragB = ((lane >> 4) << 8) + ((lane & 15) << 4);

  v8f acc[4];
#pragma unroll
  for (int i = 0; i < 4; ++i)
#pragma unroll
    for (int r = 0; r < 8; ++r) acc[i][r] = 0.f;
  __syncthreads();                              // kdec visible

  for (int k0 = 0; k0 < K3; k0 += 32) {
    {                                           // async weight block (4 KB)
      async_cp16((uint4*)Bs + t,
                 (const uint4*)(B3w + ((size_t)(k0 >> 5) * 4) * 512) + t);
    }
    float vals[16];
#pragma unroll
    for (int it = 0; it < 16; ++it) {           // branchless batched im2col
      int kk = kkb + it * 2;
      int d = kdec[k0 + kk];
      int c = d >> 4;
      int hh = hq + ((d >> 2) & 3) - 1;
      int ww = wq + (d & 3) - 1;
      bool ok = ((unsigned)hh < H_IMG) && ((unsigned)ww < W_IMG);
      int off = c * L_PIX + (ok ? (hh * W_IMG + ww) : 0);
      float v = xin[off];
      vals[it] = ok ? v : 0.f;
    }
#pragma unroll
    for (int it = 0; it < 16; ++it)
      As[m * 32 + kkb + it * 2] = (half_t)vals[it];
    async_wait_all();
    __syncthreads();
    v16h a = ld_frag_a(As + wave * 16 * 32, 32, lane);
#pragma unroll
    for (int nt = 0; nt < 4; ++nt) {
      v16h b = ld_frag_ctg(Bs + nt * 512 + fragB);
      acc[nt] = wmma32(a, b, acc[nt]);
    }
    __syncthreads();
  }
  int col = lane & 15, rofs = (lane >> 4) * 8;
  float* orow = xemb + ((size_t)n * L_PIX + l0 + wave * 16) * C_R;
#pragma unroll
  for (int nt = 0; nt < 4; ++nt) {
    int c = nt * 16 + col;
    float bv = bias[c];
#pragma unroll
    for (int r = 0; r < 8; ++r)
      orow[(size_t)(rofs + r) * C_R + c] = acc[nt][r] + bv;
  }
}

// ---------------- kernel 2: 1x1 conv GEMM (WMMA), out f16 ---------------------
// per WG: 128 pixels x 256 channels, K = 256. 8 waves x (16 rows, 16 N-tiles).

__global__ __launch_bounds__(256) void conv1x1_wmma(const float* __restrict__ x,
                                                    const half_t* __restrict__ B1w,
                                                    const float* __restrict__ bias,
                                                    half_t* __restrict__ yemb) {
  __shared__ half_t As[128 * 32];
  __shared__ half_t Bs[16 * 512];   // 16 swizzled (32k x 16n) blocks
  int n  = blockIdx.x / (L_PIX / 128);
  int l0 = (blockIdx.x % (L_PIX / 128)) * 128;
  int t = threadIdx.x, wave = t >> 5, lane = t & 31;
  const float* xin = x + (size_t)n * C_IN * L_PIX + l0;
  int m = t & 127, kkb = t >> 7;
  int fragB = ((lane >> 4) << 8) + ((lane & 15) << 4);

  v8f acc[16];
#pragma unroll
  for (int i = 0; i < 16; ++i)
#pragma unroll
    for (int r = 0; r < 8; ++r) acc[i][r] = 0.f;

  for (int k0 = 0; k0 < C_IN; k0 += 32) {
#pragma unroll
    for (int it = 0; it < 4; ++it) {            // async weight blocks (16 KB)
      int e = t + it * 256;
      async_cp16((uint4*)Bs + e,
                 (const uint4*)(B1w + ((size_t)(k0 >> 5) * 16) * 512) + e);
    }
    float vals[16];
#pragma unroll
    for (int it = 0; it < 16; ++it)
      vals[it] = xin[(size_t)(k0 + kkb + it * 2) * L_PIX + m];
#pragma unroll
    for (int it = 0; it < 16; ++it)
      As[m * 32 + kkb + it * 2] = (half_t)vals[it];
    async_wait_all();
    __syncthreads();
    v16h a = ld_frag_a(As + wave * 16 * 32, 32, lane);
#pragma unroll
    for (int nt = 0; nt < 16; ++nt) {
      v16h b = ld_frag_ctg(Bs + nt * 512 + fragB);
      acc[nt] = wmma32(a, b, acc[nt]);
    }
    __syncthreads();
  }
  int col = lane & 15, rofs = (lane >> 4) * 8;
  half_t* orow = yemb + ((size_t)n * L_PIX + l0 + wave * 16) * C_IN;
#pragma unroll
  for (int nt = 0; nt < 16; ++nt) {
    int c = nt * 16 + col;
    float bv = bias[c];
#pragma unroll
    for (int r = 0; r < 8; ++r)
      orow[(size_t)(rofs + r) * C_IN + c] = (half_t)(acc[nt][r] + bv);
  }
}

// ---------------- kernel 3: l2norm + argmax over 128 means --------------------

__global__ __launch_bounds__(256) void assign_clusters(const float* __restrict__ xemb,
                                                       const float* __restrict__ means,
                                                       int* __restrict__ codes) {
  __shared__ float sM[NCLUST * C_R];          // 32 KB
  int t = threadIdx.x;
  for (int e = t; e < NCLUST * C_R; e += 256) sM[e] = means[e];
  __syncthreads();
  int gp = blockIdx.x * 256 + t;
  const float* row = xemb + (size_t)gp * C_R;
  float v[C_R];
  float ss = 0.f;
#pragma unroll
  for (int c = 0; c < C_R; ++c) { v[c] = row[c]; ss += v[c] * v[c]; }
  float inv = 1.f / fmaxf(sqrtf(ss), 1e-12f);
#pragma unroll
  for (int c = 0; c < C_R; ++c) v[c] *= inv;
  float best = -3.402e38f; int bi = 0;
  for (int cl = 0; cl < NCLUST; ++cl) {
    float d = 0.f;
#pragma unroll
    for (int c = 0; c < C_R; ++c) d += v[c] * sM[cl * C_R + c];
    if (d > best) { best = d; bi = cl; }
  }
  codes[gp] = bi;
}

// ---------------- kernels 4-6: stable counting sort (128 buckets) -------------

__global__ __launch_bounds__(256) void hist_kernel(const int* __restrict__ codes,
                                                   int* __restrict__ hist) {
  __shared__ int h[NCLUST];
  int t = threadIdx.x;
  if (t < NCLUST) h[t] = 0;
  __syncthreads();
  atomicAdd(&h[codes[blockIdx.x * 256 + t]], 1);
  __syncthreads();
  if (t < NCLUST) hist[blockIdx.x * NCLUST + t] = h[t];
}

__global__ __launch_bounds__(128) void scan_kernel(const int* __restrict__ hist,
                                                   int* __restrict__ base) {
  __shared__ int tot[NCLUST];
  int n = blockIdx.x, b = threadIdx.x;
  const int* hn = hist + (size_t)n * NCHUNK * NCLUST;
  int run = 0;
  for (int c = 0; c < NCHUNK; ++c) run += hn[c * NCLUST + b];
  tot[b] = run;
  __syncthreads();
  for (int off = 1; off < NCLUST; off <<= 1) {   // inclusive scan
    int xv = (b >= off) ? tot[b - off] : 0;
    __syncthreads();
    tot[b] += xv;
    __syncthreads();
  }
  int run2 = tot[b] - run;                        // exclusive bucket start
  int* bs = base + ((size_t)n * NCLUST + b) * NCHUNK;
  for (int c = 0; c < NCHUNK; ++c) { bs[c] = run2; run2 += hn[c * NCLUST + b]; }
}

__global__ __launch_bounds__(256) void scatter_kernel(const int* __restrict__ codes,
                                                      const int* __restrict__ base,
                                                      int* __restrict__ indices,
                                                      int* __restrict__ undo) {
  __shared__ int sc[256];
  int n = blockIdx.x / NCHUNK, ch = blockIdx.x % NCHUNK;
  int t = threadIdx.x;
  int il = ch * 256 + t;
  int myb = codes[n * L_PIX + il];
  sc[t] = myb;
  __syncthreads();
  int rank = 0;
  for (int j = 0; j < t; ++j) rank += (sc[j] == myb) ? 1 : 0;   // stable rank
  int p = base[((size_t)n * NCLUST + myb) * NCHUNK + ch] + rank;
  indices[n * L_PIX + p] = il;
  undo[n * L_PIX + il] = p;
}

// ---------------- kernel 7: gather sorted xs/xn/ys (f16) ----------------------

__global__ __launch_bounds__(256) void gather_sorted(const float* __restrict__ xemb,
                                                     const half_t* __restrict__ yemb,
                                                     const int* __restrict__ indices,
                                                     half_t* __restrict__ xs,
                                                     half_t* __restrict__ xn,
                                                     half_t* __restrict__ ys) {
  __shared__ int   sidx[64];
  __shared__ float sinv[64];
  int n  = blockIdx.x / (L_PIX / 64);
  int p0 = (blockIdx.x % (L_PIX / 64)) * 64;
  int t = threadIdx.x;
  if (t < 64) {
    int i = indices[n * L_PIX + p0 + t];
    sidx[t] = i;
    const float* row = xemb + ((size_t)n * L_PIX + i) * C_R;
    float ss = 0.f;
    for (int c = 0; c < C_R; ++c) { float v = row[c]; ss += v * v; }
    sinv[t] = 1.f / fmaxf(sqrtf(ss), 5e-5f);    // key-norm eps from reference
  }
  __syncthreads();
#pragma unroll
  for (int it = 0; it < 16; ++it) {             // 64 pos x 64 ch
    int e = t + it * 256;
    int q = e >> 6, c = e & 63;
    float v = xemb[((size_t)n * L_PIX + sidx[q]) * C_R + c];
    size_t o = ((size_t)n * L_PIX + p0 + q) * C_R + c;
    xs[o] = (half_t)v;
    xn[o] = (half_t)(v * sinv[q]);
  }
  const uint4* ysrc = (const uint4*)(yemb + (size_t)n * L_PIX * C_IN);
  uint4* ydst = (uint4*)(ys + ((size_t)n * L_PIX + p0) * C_IN);
#pragma unroll
  for (int it = 0; it < 8; ++it) {              // 64 rows x 512B
    int e = t + it * 256;
    int q = e >> 5, w = e & 31;
    ydst[q * 32 + w] = ysrc[(size_t)sidx[q] * 32 + w];
  }
}

// ---------------- kernel 8: windowed flash attention (WMMA) -------------------
// one WG per (n, window); 9 waves, each owns a 16-row query tile.
// j-space = 432 keys (win, win-1, win+1); online softmax; V streamed in 32-row
// chunks, staged into fragment-ready swizzled LDS blocks.

__global__ __launch_bounds__(ATTN_THREADS) void attn_kernel(const half_t* __restrict__ xs,
                                                            const half_t* __restrict__ xn,
                                                            const half_t* __restrict__ ys,
                                                            float* __restrict__ ret) {
  extern __shared__ half_t sm[];
  half_t* sQ = sm;                      // 144*64
  half_t* sK = sQ + WIN * C_R;          // 432*64
  half_t* sV = sK + 3 * WIN * C_R;      // 32*256, swizzled [(kt*16+nt)*16+nl][i]
  half_t* sP = sV + 32 * C_IN;          // 9*16*32
  int n = blockIdx.x / KWIN, win = blockIdx.x % KWIN;
  int t = threadIdx.x, wave = t >> 5, lane = t & 31;
  size_t nbase = (size_t)n * L_PIX;
  int wprev = (win + KWIN - 1) & (KWIN - 1);
  int wnext = (win + 1) & (KWIN - 1);

  const uint4* xs4 = (const uint4*)(xs + (nbase + (size_t)win * WIN) * C_R);
  uint4* sQ4 = (uint4*)sQ;
  for (int e = t; e < WIN * 8; e += ATTN_THREADS)
    async_cp16(sQ4 + e, xs4 + e);
  uint4* sK4 = (uint4*)sK;
  const uint4* xn4 = (const uint4*)xn;
  for (int e = t; e < 3 * WIN * 8; e += ATTN_THREADS) {
    int j = e >> 3, w = e & 7;
    int seg = j / WIN, jj = j - seg * WIN;
    int wsrc = (seg == 0) ? win : ((seg == 1) ? wprev : wnext);
    async_cp16(sK4 + e, xn4 + ((nbase + (size_t)wsrc * WIN + jj) * C_R >> 3) + w);
  }

  float mrun[8], lrun[8];
  v8f acc[16];
#pragma unroll
  for (int r = 0; r < 8; ++r) { mrun[r] = -1e30f; lrun[r] = 0.f; }
#pragma unroll
  for (int i = 0; i < 16; ++i)
#pragma unroll
    for (int r = 0; r < 8; ++r) acc[i][r] = 0.f;
  async_wait_all();
  __syncthreads();

  const half_t* qbase = sQ + wave * 16 * C_R;
  half_t* pw = sP + wave * 16 * 32;
  const int njt = 3 * WIN / 16;                 // 27 key tiles
  const uint4* ys4 = (const uint4*)ys;
  int fragV = ((lane >> 4) << 12) + ((lane & 15) << 4);

  for (int ch = 0; ch < 14; ++ch) {             // 14 chunks of 32 j (last half-full)
    for (int e = t; e < 32 * 32; e += ATTN_THREADS) {
      int jj = e >> 5, w = e & 31;
      int j = ch * 32 + jj;
      bool ok = j < 3 * WIN;
      int seg = (j >= 2 * WIN) ? 2 : ((j >= WIN) ? 1 : 0);
      int jo = ok ? (j - seg * WIN) : 0;
      int wsrc = (seg == 0) ? win : ((seg == 1) ? wprev : wnext);
      if (!ok) wsrc = win;
      uint4 val = ys4[((nbase + (size_t)wsrc * WIN + jo) * C_IN >> 3) + w];
      if (!ok) val = make_uint4(0u, 0u, 0u, 0u); // softmax tail rows
      v8h hv = __builtin_bit_cast(v8h, val);
      int kt = jj >> 4, i = jj & 15, c0 = w * 8;
#pragma unroll
      for (int q = 0; q < 8; ++q) {             // transpose into swizzled layout
        int c = c0 + q;
        sV[(((kt * 16 + (c >> 4)) * 16) + (c & 15)) * 16 + i] = hv[q];
      }
    }
    if (t < 32 && ch + 1 < 14) {                // prefetch next V chunk row starts
      int j = (ch + 1) * 32 + t;
      if (j < 3 * WIN) {
        int seg = j / WIN, jo = j - seg * WIN;
        int wsrc = (seg == 0) ? win : ((seg == 1) ? wprev : wnext);
        __builtin_prefetch(ys + (nbase + (size_t)wsrc * WIN + jo) * C_IN, 0, 0);
      }
    }
    __syncthreads();

    int jt0 = ch * 2, jt1 = ch * 2 + 1;
    bool v1 = (jt1 < njt);                      // uniform: EXEC stays all-ones
    v8f s0, s1;
#pragma unroll
    for (int r = 0; r < 8; ++r) { s0[r] = 0.f; s1[r] = 0.f; }
#pragma unroll
    for (int kk = 0; kk < 2; ++kk) {            // K = 64 in two 32-steps
      v16h a = ld_frag_a(qbase + kk * 32, C_R, lane);
      v16h b0 = ld_frag_bT(sK + (size_t)jt0 * 16 * C_R + kk * 32, C_R, lane);
      s0 = wmma32(a, b0, s0);
      if (v1) {
        v16h b1 = ld_frag_bT(sK + (size_t)jt1 * 16 * C_R + kk * 32, C_R, lane);
        s1 = wmma32(a, b1, s1);
      }
    }

    float f[8];
#pragma unroll
    for (int r = 0; r < 8; ++r) {               // row max across 16 lanes
      float m = s0[r];
      if (v1) m = fmaxf(m, s1[r]);
#pragma unroll
      for (int msk = 8; msk >= 1; msk >>= 1) m = fmaxf(m, __shfl_xor(m, msk, 32));
      float mn = fmaxf(mrun[r], m);
      f[r] = __expf(mrun[r] - mn);
      mrun[r] = mn;
    }
    v8f p0, p1;
#pragma unroll
    for (int r = 0; r < 8; ++r) {
      p0[r] = __expf(s0[r] - mrun[r]);
      p1[r] = v1 ? __expf(s1[r] - mrun[r]) : 0.f;
      float s = p0[r] + p1[r];
#pragma unroll
      for (int msk = 8; msk >= 1; msk >>= 1) s += __shfl_xor(s, msk, 32);
      lrun[r] = lrun[r] * f[r] + s;
    }
    {                                           // D-layout -> A-layout via LDS
      int col = lane & 15, rofs = (lane >> 4) * 8;
#pragma unroll
      for (int r = 0; r < 8; ++r) {
        pw[(rofs + r) * 32 + col] = (half_t)p0[r];
        pw[(rofs + r) * 32 + 16 + col] = (half_t)p1[r];
      }
    }
    v16h aP = ld_frag_a(pw, 32, lane);          // same-wave LDS: DS ops in-order
#pragma unroll
    for (int nt = 0; nt < 16; ++nt) {
#pragma unroll
      for (int r = 0; r < 8; ++r) acc[nt][r] *= f[r];
      v16h bV = ld_frag_ctg(sV + (nt << 8) + fragV);
      acc[nt] = wmma32(aP, bV, acc[nt]);
    }
    __syncthreads();
  }

  int col = lane & 15, rofs = (lane >> 4) * 8;
  float* orow = ret + (nbase + (size_t)win * WIN + wave * 16) * C_IN;
#pragma unroll
  for (int nt = 0; nt < 16; ++nt) {
    int c = nt * 16 + col;
#pragma unroll
    for (int r = 0; r < 8; ++r)
      orow[(size_t)(rofs + r) * C_IN + c] = acc[nt][r] / lrun[r];
  }
}

// ---------------- kernel 9: unsort + residual, LDS-tiled transpose ------------

__global__ __launch_bounds__(256) void unsort_out(const float* __restrict__ ret,
                                                  const int* __restrict__ undo,
                                                  const float* __restrict__ xin,
                                                  float* __restrict__ out) {
  __shared__ float tile[64][65];
  __shared__ int su[64];
  int bi = blockIdx.x;
  int n = bi / (4 * 144);
  int rem = bi - n * (4 * 144);
  int cb = rem / 144, i0 = (rem - cb * 144) * 64;
  int t = threadIdx.x;
  if (t < 64) su[t] = undo[n * L_PIX + i0 + t];
  __syncthreads();
#pragma unroll
  for (int it = 0; it < 16; ++it) {             // gather: 256B contiguous per row
    int e = t + it * 256;
    int ii = e >> 6, c = e & 63;
    tile[ii][c] = ret[((size_t)n * L_PIX + su[ii]) * C_IN + cb * 64 + c];
  }
  __syncthreads();
#pragma unroll
  for (int it = 0; it < 16; ++it) {             // coalesced NCHW stores
    int e = t + it * 256;
    int c = e >> 6, ii = e & 63;
    size_t o = ((size_t)n * C_IN + cb * 64 + c) * L_PIX + i0 + ii;
    out[o] = xin[o] + 0.1f * tile[ii][c];
  }
}

// ---------------- host launcher ----------------------------------------------

extern "C" void kernel_launch(void* const* d_in, const int* in_sizes, int n_in,
                              void* d_out, int out_size, void* d_ws, size_t ws_size,
                              hipStream_t stream) {
  const float* input_x = (const float*)d_in[0];
  const float* w_match = (const float*)d_in[1];
  const float* b_match = (const float*)d_in[2];
  const float* w_asm   = (const float*)d_in[3];
  const float* b_asm   = (const float*)d_in[4];
  const float* means   = (const float*)d_in[5];
  float* out = (float*)d_out;

  char* ws = (char*)d_ws;
  auto carve = [&](size_t bytes) {
    char* p = ws;
    ws += (bytes + 255) & ~(size_t)255;
    return p;
  };
  half_t* B3   = (half_t*)carve((size_t)K3 * C_R * sizeof(half_t));
  half_t* B1   = (half_t*)carve((size_t)C_IN * C_IN * sizeof(half_t));
  float*  xemb = (float*) carve((size_t)N_B * L_PIX * C_R * sizeof(float));
  half_t* yemb = (half_t*)carve((size_t)N_B * L_PIX * C_IN * sizeof(half_t));
  int*    codes = (int*)  carve((size_t)N_B * L_PIX * sizeof(int));
  int*    hist  = (int*)  carve((size_t)N_B * NCHUNK * NCLUST * sizeof(int));
  int*    base  = (int*)  carve((size_t)N_B * NCLUST * NCHUNK * sizeof(int));
  int*    indices = (int*)carve((size_t)N_B * L_PIX * sizeof(int));
  int*    undo    = (int*)carve((size_t)N_B * L_PIX * sizeof(int));
  half_t* xsrt = (half_t*)carve((size_t)N_B * L_PIX * C_R * sizeof(half_t));
  half_t* xnrm = (half_t*)carve((size_t)N_B * L_PIX * C_R * sizeof(half_t));
  half_t* ysrt = (half_t*)carve((size_t)N_B * L_PIX * C_IN * sizeof(half_t));
  float*  retb = (float*) carve((size_t)N_B * L_PIX * C_IN * sizeof(float));

  int prepN = (K3 * C_R + C_IN * C_IN + 255) / 256;
  prep_weights<<<prepN, 256, 0, stream>>>(w_match, w_asm, B3, B1);
  conv3x3_wmma<<<N_B * (L_PIX / 128), 256, 0, stream>>>(input_x, B3, b_match, xemb);
  conv1x1_wmma<<<N_B * (L_PIX / 128), 256, 0, stream>>>(input_x, B1, b_asm, yemb);
  assign_clusters<<<N_B * L_PIX / 256, 256, 0, stream>>>(xemb, means, codes);
  hist_kernel<<<N_B * NCHUNK, 256, 0, stream>>>(codes, hist);
  scan_kernel<<<N_B, 128, 0, stream>>>(hist, base);
  scatter_kernel<<<N_B * NCHUNK, 256, 0, stream>>>(codes, base, indices, undo);
  gather_sorted<<<N_B * (L_PIX / 64), 256, 0, stream>>>(xemb, yemb, indices,
                                                        xsrt, xnrm, ysrt);
  size_t attnLds = (size_t)(WIN * C_R + 3 * WIN * C_R + 32 * C_IN + 9 * 16 * 32) *
                   sizeof(half_t);               // ~99 KB of the 320 KB WGP LDS
  attn_kernel<<<N_B * KWIN, ATTN_THREADS, attnLds, stream>>>(xsrt, xnrm, ysrt, retb);
  unsort_out<<<N_B * 4 * 144, 256, 0, stream>>>(retb, undo, input_x, out);
}
